// LRU_65652870086889
// MI455X (gfx1250) — matrix operand order repaired and checked by reading, not compiled, not run on hardware
//
#include <hip/hip_runtime.h>

// ---------------------------------------------------------------------------
// LRU forward for MI455X (gfx1250): bf16 WMMA GEMMs with async global->LDS
// double-buffered pipeline + fp32 chunked scan.
// ---------------------------------------------------------------------------

#define T_LEN   16384
#define IN_D    1024
#define H_D     1024
#define OUT_D   1024
#define CHUNK   256
#define NCHUNK  (T_LEN / CHUNK)   // 64
#define KBIG    (3 * H_D)         // 3072 fused K for the output GEMM

typedef __attribute__((ext_vector_type(16))) __bf16 v16bf;
typedef __attribute__((ext_vector_type(8)))  float  v8f;
typedef __attribute__((__vector_size__(16))) int    vi4;   // builtin's b128 type

#define AS1 __attribute__((address_space(1)))
#define AS3 __attribute__((address_space(3)))

union FragU { uint4 q[2]; v16bf v; };

__device__ __forceinline__ unsigned short f2bf(float f) {
  unsigned int u = __float_as_uint(f);
  if ((u & 0x7fffffffu) > 0x7f800000u) return (unsigned short)0x7fc0; // NaN
  unsigned int r = u + 0x7fffu + ((u >> 16) & 1u);                    // RNE
  return (unsigned short)(r >> 16);
}

// --------------------------- parameter prep -------------------------------

__global__ void lru_lambda_kernel(const float* __restrict__ nu_log,
                                  const float* __restrict__ theta_log,
                                  float* __restrict__ lam_re,
                                  float* __restrict__ lam_im) {
  int h = blockIdx.x * blockDim.x + threadIdx.x;
  if (h >= H_D) return;
  float mag = __expf(-__expf(nu_log[h]));
  float th  = __expf(theta_log[h]);
  lam_re[h] = mag * __cosf(th);
  lam_im[h] = mag * __sinf(th);
}

// Bn_t[n][i], n in [0,2H): pre-transposed [N][K] bf16 weight for GEMM1.
// Faithful to reference quirk: gamma_log is indexed by the *input* dim i.
__global__ void lru_bnorm_kernel(const float* __restrict__ B_re,
                                 const float* __restrict__ B_im,
                                 const float* __restrict__ gamma_log,
                                 unsigned short* __restrict__ Bn_t) {
  int idx = blockIdx.x * blockDim.x + threadIdx.x;  // over 2H * IN
  if (idx >= 2 * H_D * IN_D) return;
  int i = idx % IN_D;
  int n = idx / IN_D;
  float g = __expf(gamma_log[i]);
  float v = (n < H_D) ? B_re[(size_t)i * H_D + n]
                      : B_im[(size_t)i * H_D + (n - H_D)];
  Bn_t[(size_t)n * IN_D + i] = f2bf(v * g);
}

// W_t[o][k], k in [0,3H): [C_re(o,h) | -C_im(o,h) | D('io' layout)] bf16.
__global__ void lru_wbig_kernel(const float* __restrict__ C_re,
                                const float* __restrict__ C_im,
                                const float* __restrict__ D,
                                unsigned short* __restrict__ W_t) {
  int idx = blockIdx.x * blockDim.x + threadIdx.x;  // over OUT * 3H
  if (idx >= OUT_D * KBIG) return;
  int k = idx % KBIG;
  int o = idx / KBIG;
  float v;
  if (k < H_D)            v =  C_re[(size_t)o * H_D + k];
  else if (k < 2 * H_D)   v = -C_im[(size_t)o * H_D + (k - H_D)];
  else                    v =  D[(size_t)(k - 2 * H_D) * OUT_D + o]; // D as 'io'
  W_t[(size_t)o * KBIG + k] = f2bf(v);
}

// X -> bf16 into the fused A matrix, columns [2H, 3H).
__global__ void lru_xconv_kernel(const float* __restrict__ x,
                                 unsigned short* __restrict__ A_big) {
  size_t idx = (size_t)blockIdx.x * blockDim.x + threadIdx.x; // T*IN
  if (idx >= (size_t)T_LEN * IN_D) return;
  size_t t = idx / IN_D;
  int    i = (int)(idx % IN_D);
  A_big[t * KBIG + 2 * H_D + i] = f2bf(x[idx]);
}

// ------------------------------- scan -------------------------------------
// Pass A: local scan inside each chunk (zero initial state), in place on Bu.
__global__ void lru_scan_local_kernel(float* __restrict__ Bu,
                                      const float* __restrict__ lam_re,
                                      const float* __restrict__ lam_im,
                                      float* __restrict__ cf_re,
                                      float* __restrict__ cf_im) {
  int h = blockIdx.x * blockDim.x + threadIdx.x;   // 0..H-1
  int c = blockIdx.y;                              // 0..NCHUNK-1
  float lre = lam_re[h], lim = lam_im[h];
  float hre = 0.f, him = 0.f;
  size_t base = (size_t)c * CHUNK * (2 * H_D) + h;
  for (int j = 0; j < CHUNK; ++j) {
    float ure = Bu[base];
    float uim = Bu[base + H_D];
    float nre = __fmaf_rn(lre, hre, __fmaf_rn(-lim, him, ure));
    float nim = __fmaf_rn(lre, him, __fmaf_rn( lim, hre, uim));
    hre = nre; him = nim;
    Bu[base]       = hre;
    Bu[base + H_D] = him;
    base += 2 * H_D;
  }
  cf_re[(size_t)c * H_D + h] = hre;
  cf_im[(size_t)c * H_D + h] = him;
}

// Pass B: exclusive scan of chunk carries per channel (uses lambda^CHUNK).
__global__ void lru_scan_carry_kernel(const float* __restrict__ lam_re,
                                      const float* __restrict__ lam_im,
                                      const float* __restrict__ cf_re,
                                      const float* __restrict__ cf_im,
                                      float* __restrict__ ex_re,
                                      float* __restrict__ ex_im) {
  int h = blockIdx.x * blockDim.x + threadIdx.x;
  if (h >= H_D) return;
  float lre = lam_re[h], lim = lam_im[h];
  float pre = 1.f, pim = 0.f;                 // lambda^CHUNK
  for (int j = 0; j < CHUNK; ++j) {
    float nre = pre * lre - pim * lim;
    float nim = pre * lim + pim * lre;
    pre = nre; pim = nim;
  }
  float are = 0.f, aim = 0.f;
  for (int c = 0; c < NCHUNK; ++c) {
    ex_re[(size_t)c * H_D + h] = are;
    ex_im[(size_t)c * H_D + h] = aim;
    float fre = cf_re[(size_t)c * H_D + h];
    float fim = cf_im[(size_t)c * H_D + h];
    float nre = pre * are - pim * aim + fre;
    float nim = pre * aim + pim * are + fim;
    are = nre; aim = nim;
  }
}

// Pass C: h_t = h_local_t + lambda^(j+1) * carry; fused bf16 cast into A_big.
__global__ void lru_scan_apply_kernel(const float* __restrict__ Bu,
                                      const float* __restrict__ lam_re,
                                      const float* __restrict__ lam_im,
                                      const float* __restrict__ ex_re,
                                      const float* __restrict__ ex_im,
                                      unsigned short* __restrict__ A_big) {
  int h = blockIdx.x * blockDim.x + threadIdx.x;
  int c = blockIdx.y;
  float lre = lam_re[h], lim = lam_im[h];
  float cre = ex_re[(size_t)c * H_D + h];
  float cim = ex_im[(size_t)c * H_D + h];
  float pre = lre, pim = lim;                 // lambda^1
  size_t bu = (size_t)c * CHUNK * (2 * H_D) + h;
  size_t ab = (size_t)c * CHUNK * KBIG + h;
  for (int j = 0; j < CHUNK; ++j) {
    float hre = Bu[bu]       + (pre * cre - pim * cim);
    float him = Bu[bu + H_D] + (pre * cim + pim * cre);
    A_big[ab]       = f2bf(hre);
    A_big[ab + H_D] = f2bf(him);
    float nre = pre * lre - pim * lim;
    float nim = pre * lim + pim * lre;
    pre = nre; pim = nim;
    bu += 2 * H_D;
    ab += KBIG;
  }
}

// ------------------------------- GEMM -------------------------------------
// C(MxN, f32) = A(MxK, bf16, row-major, pitch lda) @ Bt(NxK, bf16)^T
// 128x128x32 block tile, 4 wave32 waves (2x2), each wave owns a 64x64 tile
// -> 16 WMMA per wave per K-step from 8 fragment loads (good LDS reuse).
// Double-buffered LDS filled with GLOBAL_LOAD_ASYNC_TO_LDS_B128 (ASYNCcnt).

#define BM 128
#define BN 128
#define BK 32
#define LDP 40   // LDS pitch in halfs (80 B = 20 banks -> conflict-free)

#define WG_BARRIER() \
  asm volatile("s_barrier_signal -1\n\ts_barrier_wait -1" ::: "memory")
// Trailing barrier: drain this wave's LDS reads BEFORE signaling, so the next
// iteration's async writes cannot overwrite LDS still being read.
#define WG_BARRIER_DS() \
  asm volatile("s_wait_dscnt 0x0\n\ts_barrier_signal -1\n\ts_barrier_wait -1" \
               ::: "memory")

#define GLD_ASYNC_B128(gp, lp) \
  __builtin_amdgcn_global_load_async_to_lds_b128( \
      (AS1 vi4*)(vi4*)(gp), (AS3 vi4*)(vi4*)(lp), 0, 0)

__global__ __launch_bounds__(128)
void lru_gemm_bf16_kernel(const unsigned short* __restrict__ A, int lda,
                          const unsigned short* __restrict__ Bt,
                          float* __restrict__ Cout, int N, int K) {
  __shared__ __align__(16) unsigned short lA[2][BM * LDP];
  __shared__ __align__(16) unsigned short lB[2][BN * LDP];

  const int tid  = threadIdx.x;
  const int lane = tid & 31;
  const int wid  = tid >> 5;
  const int wm   = wid >> 1;            // 0..1 -> 64 rows
  const int wn   = wid & 1;             // 0..1 -> 64 cols
  const int bm0  = blockIdx.y * BM;
  const int bn0  = blockIdx.x * BN;
  const int ln16 = lane & 15;
  const int hiA  = (lane >> 4) * 8;     // A frag: K halves offset {0,8}
  const int hiB  = (lane >> 4) * 8;     // B frag: K halves offset {0,8}
  const int ldb  = K;

  // Staging: 128 threads, each covers rows r+32s (s=0..3), one 16B chunk.
  const int r  = tid >> 2;              // 0..31
  const int cc = (tid & 3) * 8;         // K-halves offset {0,8,16,24}
  const unsigned short* gA[4];
  const unsigned short* gB[4];
  unsigned short* sA[4];
  unsigned short* sB[4];
#pragma unroll
  for (int s = 0; s < 4; ++s) {
    gA[s] = &A[(size_t)(bm0 + r + 32 * s) * lda + cc];
    gB[s] = &Bt[(size_t)(bn0 + r + 32 * s) * ldb + cc];
    sA[s] = &lA[0][(r + 32 * s) * LDP + cc];
    sB[s] = &lB[0][(r + 32 * s) * LDP + cc];
  }
  const size_t bufStride = (size_t)BM * LDP;  // halfs between buffer 0 and 1

  v8f zero = {0.f, 0.f, 0.f, 0.f, 0.f, 0.f, 0.f, 0.f};
  v8f acc[4][4];
#pragma unroll
  for (int mt = 0; mt < 4; ++mt)
#pragma unroll
    for (int nt = 0; nt < 4; ++nt) acc[mt][nt] = zero;

  // Issue one tile's async copies (8 x b128 per thread, tracked by ASYNCcnt).
  auto issue_tile = [&](int kt, int buf) {
    size_t so = (size_t)buf * bufStride;
#pragma unroll
    for (int s = 0; s < 4; ++s) {
      GLD_ASYNC_B128(gA[s] + kt, sA[s] + so);
      GLD_ASYNC_B128(gB[s] + kt, sB[s] + so);
    }
  };

  // Consume one LDS-resident tile: 8 fragment loads, 16 WMMAs.
  auto compute_tile = [&](int buf) {
    FragU af[4], bf2[4];
#pragma unroll
    for (int mt = 0; mt < 4; ++mt) {
      const unsigned short* p = &lA[buf][(wm * 64 + mt * 16 + ln16) * LDP + hiA];
      af[mt].q[0] = *(const uint4*)p;          // K = hiA .. hiA+7
      af[mt].q[1] = *(const uint4*)(p + 16);   // K = hiA+16 .. hiA+23
    }
#pragma unroll
    for (int nt = 0; nt < 4; ++nt) {
      const unsigned short* p = &lB[buf][(wn * 64 + nt * 16 + ln16) * LDP + hiB];
      bf2[nt].q[0] = *(const uint4*)p;         // K = hiB .. hiB+7
      bf2[nt].q[1] = *(const uint4*)(p + 16);  // K = hiB+16 .. hiB+23
    }
#pragma unroll
    for (int mt = 0; mt < 4; ++mt)
#pragma unroll
      for (int nt = 0; nt < 4; ++nt)
        acc[mt][nt] = __builtin_amdgcn_wmma_f32_16x16x32_bf16(
            false, af[mt].v, false, bf2[nt].v, (short)0, acc[mt][nt],
            false, false);
  };

  const int nk = K / BK;
  issue_tile(0, 0);

  for (int i = 0; i < nk - 1; ++i) {        // steady state: branch-free waits
    const int buf = i & 1;
    issue_tile((i + 1) * BK, buf ^ 1);      // prefetch next tile
    __builtin_amdgcn_s_wait_asynccnt(8);    // tile i's 8 asyncs retired
    WG_BARRIER();                           // whole tile i visible in LDS
    compute_tile(buf);
    WG_BARRIER_DS();                        // reads drained -> buf reusable
  }
  __builtin_amdgcn_s_wait_asynccnt(0);      // last tile resident
  WG_BARRIER();
  compute_tile((nk - 1) & 1);

  // Epilogue: lane N = lane%16, VGPR v holds M = v + (lane>=16 ? 8 : 0).
  const int hi8 = (lane >> 4) * 8;
#pragma unroll
  for (int mt = 0; mt < 4; ++mt)
#pragma unroll
    for (int nt = 0; nt < 4; ++nt) {
      int row0 = bm0 + wm * 64 + mt * 16 + hi8;
      int col  = bn0 + wn * 64 + nt * 16 + ln16;
#pragma unroll
      for (int v = 0; v < 8; ++v)
        Cout[(size_t)(row0 + v) * N + col] = acc[mt][nt][v];
    }
}

// ----------------------------- launcher -----------------------------------

extern "C" void kernel_launch(void* const* d_in, const int* in_sizes, int n_in,
                              void* d_out, int out_size, void* d_ws, size_t ws_size,
                              hipStream_t stream) {
  (void)in_sizes; (void)n_in; (void)out_size; (void)ws_size;
  const float* x         = (const float*)d_in[0];
  const float* nu_log    = (const float*)d_in[1];
  const float* theta_log = (const float*)d_in[2];
  const float* gamma_log = (const float*)d_in[3];
  const float* B_re      = (const float*)d_in[4];
  const float* B_im      = (const float*)d_in[5];
  const float* C_re      = (const float*)d_in[6];
  const float* C_im      = (const float*)d_in[7];
  const float* D         = (const float*)d_in[8];
  float* out = (float*)d_out;

  // Workspace layout (256 B aligned)
  char* ws = (char*)d_ws;
  size_t off = 0;
  auto alloc = [&](size_t bytes) {
    void* p = ws + off;
    off += (bytes + 255) & ~(size_t)255;
    return p;
  };
  float*          lam_re = (float*)alloc(H_D * 4);
  float*          lam_im = (float*)alloc(H_D * 4);
  unsigned short* Bn_t   = (unsigned short*)alloc((size_t)2 * H_D * IN_D * 2);  // 4 MiB
  unsigned short* W_t    = (unsigned short*)alloc((size_t)OUT_D * KBIG * 2);    // 6 MiB
  unsigned short* A_big  = (unsigned short*)alloc((size_t)T_LEN * KBIG * 2);    // 96 MiB
  float*          Bu     = (float*)alloc((size_t)T_LEN * 2 * H_D * 4);          // 128 MiB
  float*          cf_re  = (float*)alloc((size_t)NCHUNK * H_D * 4);
  float*          cf_im  = (float*)alloc((size_t)NCHUNK * H_D * 4);
  float*          ex_re  = (float*)alloc((size_t)NCHUNK * H_D * 4);
  float*          ex_im  = (float*)alloc((size_t)NCHUNK * H_D * 4);

  // 1. Parameter prep
  lru_lambda_kernel<<<(H_D + 255) / 256, 256, 0, stream>>>(nu_log, theta_log,
                                                           lam_re, lam_im);
  lru_bnorm_kernel<<<(2 * H_D * IN_D) / 256, 256, 0, stream>>>(B_re, B_im,
                                                               gamma_log, Bn_t);
  lru_wbig_kernel<<<(OUT_D * KBIG) / 256, 256, 0, stream>>>(C_re, C_im, D, W_t);
  lru_xconv_kernel<<<(int)(((size_t)T_LEN * IN_D) / 256), 256, 0, stream>>>(x, A_big);

  // 2. GEMM1: Bu(T x 2H) = X_bf16 @ [B_re_norm | B_im_norm]
  {
    dim3 grid((2 * H_D) / BN, T_LEN / BM);
    lru_gemm_bf16_kernel<<<grid, 128, 0, stream>>>(A_big + 2 * H_D, KBIG, Bn_t,
                                                   Bu, 2 * H_D, IN_D);
  }

  // 3. Chunked diagonal-complex scan (3 passes)
  {
    dim3 grid(H_D / 256, NCHUNK);
    lru_scan_local_kernel<<<grid, 256, 0, stream>>>(Bu, lam_re, lam_im,
                                                    cf_re, cf_im);
    lru_scan_carry_kernel<<<H_D / 256, 256, 0, stream>>>(lam_re, lam_im,
                                                         cf_re, cf_im,
                                                         ex_re, ex_im);
    lru_scan_apply_kernel<<<grid, 256, 0, stream>>>(Bu, lam_re, lam_im,
                                                    ex_re, ex_im, A_big);
  }

  // 4. GEMM2: out(T x O) = [h_re | h_im | X]_bf16 @ [C_reT ; -C_imT ; D]
  {
    dim3 grid(OUT_D / BN, T_LEN / BM);
    lru_gemm_bf16_kernel<<<grid, 128, 0, stream>>>(A_big, KBIG, W_t,
                                                   out, OUT_D, KBIG);
  }
}
